// LSTM_cell_10136122819072
// MI455X (gfx1250) — matrix operand order repaired and checked by reading, not compiled, not run on hardware
//
#include <hip/hip_runtime.h>
#include <hip/hip_bf16.h>

#define B_DIM  4096
#define IN_DIM 1024
#define H_DIM  1024

typedef __attribute__((ext_vector_type(16))) __bf16 v16bf;
typedef __attribute__((ext_vector_type(8)))  float  v8f;

union FragU { uint4 q[2]; v16bf v; };

__device__ __forceinline__ unsigned short f2bf(float f) {
  unsigned int u = __float_as_uint(f);
  u += 0x7FFFu + ((u >> 16) & 1u);   // round-to-nearest-even
  return (unsigned short)(u >> 16);
}

// ---------------- Phase 0: fp32 -> bf16 conversion (vectorized x4) ----------
__global__ __launch_bounds__(256) void cvt_f32_to_bf16(const float* __restrict__ src,
                                                       unsigned short* __restrict__ dst,
                                                       int n4) {
  int i = blockIdx.x * blockDim.x + threadIdx.x;
  if (i < n4) {
    float4 v = ((const float4*)src)[i];
    ushort4 o;
    o.x = f2bf(v.x); o.y = f2bf(v.y); o.z = f2bf(v.z); o.w = f2bf(v.w);
    ((ushort4*)dst)[i] = o;
  }
}

__global__ __launch_bounds__(256) void bias_sum_kernel(const float* __restrict__ bxf, const float* __restrict__ bhf,
                                                       const float* __restrict__ bxi, const float* __restrict__ bhi,
                                                       const float* __restrict__ bxo, const float* __restrict__ bho,
                                                       const float* __restrict__ bxc, const float* __restrict__ bhc,
                                                       float* __restrict__ out) {
  int j = blockIdx.x * blockDim.x + threadIdx.x;
  if (j < H_DIM) {
    out[0 * H_DIM + j] = bxf[j] + bhf[j];
    out[1 * H_DIM + j] = bxi[j] + bhi[j];
    out[2 * H_DIM + j] = bxo[j] + bho[j];
    out[3 * H_DIM + j] = bxc[j] + bhc[j];
  }
}

// ---------------- Phase 1: fused 4-gate WMMA GEMM + LSTM pointwise ----------
// A frag (16x32 bf16, MxK): lane L -> row l16; K chunks [half*8, +8) and [16+half*8, +8)
__device__ __forceinline__ v16bf load_a_frag(const unsigned short* __restrict__ row, int k, int half) {
  FragU f;
  f.q[0] = *(const uint4*)(row + k + half * 8);
  f.q[1] = *(const uint4*)(row + k + 16 + half * 8);
  return f.v;
}
// B frag (32x16 bf16, KxN): lane L -> weight row (column of W^T) n0+l16; K chunks [half*8, +8) and [16+half*8, +8)
__device__ __forceinline__ v16bf load_b_frag(const unsigned short* __restrict__ row, int k, int half) {
  FragU f;
  f.q[0] = *(const uint4*)(row + k + half * 8);
  f.q[1] = *(const uint4*)(row + k + 16 + half * 8);
  return f.v;
}

__device__ __forceinline__ float sigmoid_f(float x) { return 1.0f / (1.0f + __expf(-x)); }

#define WMMA_BF16(A, Bm, C) \
  __builtin_amdgcn_wmma_f32_16x16x32_bf16(false, (A), false, (Bm), (short)0, (C), false, false)

__global__ __launch_bounds__(256) void lstm_wmma_kernel(
    const unsigned short* __restrict__ xb, const unsigned short* __restrict__ hb,
    const unsigned short* __restrict__ Wx, const unsigned short* __restrict__ Wh,
    const float* __restrict__ bias, const float* __restrict__ c_in,
    float* __restrict__ out) {
  const int lane = threadIdx.x & 31;
  const int wave = threadIdx.x >> 5;
  const int half = lane >> 4;
  const int l16  = lane & 15;
  const int n0   = blockIdx.x * 16;               // hidden-column tile
  const int m0   = blockIdx.y * 256 + wave * 32;  // 32 batch rows per wave

  // Two A-row pointers per operand (M-subtiles m0 and m0+16)
  const unsigned short* ax0 = xb + (size_t)(m0 + l16) * IN_DIM;
  const unsigned short* ax1 = xb + (size_t)(m0 + 16 + l16) * IN_DIM;
  const unsigned short* ah0 = hb + (size_t)(m0 + l16) * H_DIM;
  const unsigned short* ah1 = hb + (size_t)(m0 + 16 + l16) * H_DIM;
  // 4 gate B pointers per operand
  const unsigned short* bx0 = Wx + (size_t)(0 * H_DIM + n0 + l16) * IN_DIM;
  const unsigned short* bx1 = Wx + (size_t)(1 * H_DIM + n0 + l16) * IN_DIM;
  const unsigned short* bx2 = Wx + (size_t)(2 * H_DIM + n0 + l16) * IN_DIM;
  const unsigned short* bx3 = Wx + (size_t)(3 * H_DIM + n0 + l16) * IN_DIM;
  const unsigned short* bh0 = Wh + (size_t)(0 * H_DIM + n0 + l16) * H_DIM;
  const unsigned short* bh1 = Wh + (size_t)(1 * H_DIM + n0 + l16) * H_DIM;
  const unsigned short* bh2 = Wh + (size_t)(2 * H_DIM + n0 + l16) * H_DIM;
  const unsigned short* bh3 = Wh + (size_t)(3 * H_DIM + n0 + l16) * H_DIM;

  v8f accf0 = {}, accf1 = {};
  v8f acci0 = {}, acci1 = {};
  v8f acco0 = {}, acco1 = {};
  v8f accc0 = {}, accc1 = {};

  // gates += x @ Wx^T   (K = IN_DIM)
#pragma unroll 2
  for (int k = 0; k < IN_DIM; k += 32) {
    v16bf a0 = load_a_frag(ax0, k, half);
    v16bf a1 = load_a_frag(ax1, k, half);
    v16bf b0 = load_b_frag(bx0, k, half);
    v16bf b1 = load_b_frag(bx1, k, half);
    v16bf b2 = load_b_frag(bx2, k, half);
    v16bf b3 = load_b_frag(bx3, k, half);
    accf0 = WMMA_BF16(a0, b0, accf0);
    accf1 = WMMA_BF16(a1, b0, accf1);
    acci0 = WMMA_BF16(a0, b1, acci0);
    acci1 = WMMA_BF16(a1, b1, acci1);
    acco0 = WMMA_BF16(a0, b2, acco0);
    acco1 = WMMA_BF16(a1, b2, acco1);
    accc0 = WMMA_BF16(a0, b3, accc0);
    accc1 = WMMA_BF16(a1, b3, accc1);
  }
  // gates += h @ Wh^T   (K = H_DIM)
#pragma unroll 2
  for (int k = 0; k < H_DIM; k += 32) {
    v16bf a0 = load_a_frag(ah0, k, half);
    v16bf a1 = load_a_frag(ah1, k, half);
    v16bf b0 = load_b_frag(bh0, k, half);
    v16bf b1 = load_b_frag(bh1, k, half);
    v16bf b2 = load_b_frag(bh2, k, half);
    v16bf b3 = load_b_frag(bh3, k, half);
    accf0 = WMMA_BF16(a0, b0, accf0);
    accf1 = WMMA_BF16(a1, b0, accf1);
    acci0 = WMMA_BF16(a0, b1, acci0);
    acci1 = WMMA_BF16(a1, b1, acci1);
    acco0 = WMMA_BF16(a0, b2, acco0);
    acco1 = WMMA_BF16(a1, b2, acco1);
    accc0 = WMMA_BF16(a0, b3, accc0);
    accc1 = WMMA_BF16(a1, b3, accc1);
  }

  // Epilogue in accumulator registers.
  // C/D layout: element r -> row (subtile_base + half*8 + r), lane -> col n0+l16.
  const int ncol = n0 + l16;
  const float bf = bias[0 * H_DIM + ncol];
  const float bi = bias[1 * H_DIM + ncol];
  const float bo = bias[2 * H_DIM + ncol];
  const float bc = bias[3 * H_DIM + ncol];
  float* __restrict__ ct_out = out;
  float* __restrict__ ht_out = out + (size_t)B_DIM * H_DIM;

#pragma unroll
  for (int sub = 0; sub < 2; ++sub) {
    const v8f& af = sub ? accf1 : accf0;
    const v8f& ai = sub ? acci1 : acci0;
    const v8f& ao = sub ? acco1 : acco0;
    const v8f& ac = sub ? accc1 : accc0;
    const int mbase = m0 + sub * 16 + half * 8;
#pragma unroll
    for (int r = 0; r < 8; ++r) {
      const size_t idx = (size_t)(mbase + r) * H_DIM + ncol;
      const float gf = sigmoid_f(af[r] + bf);
      const float gi = sigmoid_f(ai[r] + bi);
      const float go = sigmoid_f(ao[r] + bo);
      const float gc = tanhf(ac[r] + bc);
      const float cv = c_in[idx];
      const float ctv = gf * cv + gc * gi;
      const float htv = tanhf(ctv) * go;
      ct_out[idx] = ctv;
      ht_out[idx] = htv;
    }
  }
}

extern "C" void kernel_launch(void* const* d_in, const int* in_sizes, int n_in,
                              void* d_out, int out_size, void* d_ws, size_t ws_size,
                              hipStream_t stream) {
  const float* x   = (const float*)d_in[0];
  const float* c   = (const float*)d_in[1];
  const float* h   = (const float*)d_in[2];
  const float* Wxf = (const float*)d_in[3];
  const float* bxf = (const float*)d_in[4];
  const float* Whf = (const float*)d_in[5];
  const float* bhf = (const float*)d_in[6];
  const float* Wxi = (const float*)d_in[7];
  const float* bxi = (const float*)d_in[8];
  const float* Whi = (const float*)d_in[9];
  const float* bhi = (const float*)d_in[10];
  const float* Wxo = (const float*)d_in[11];
  const float* bxo = (const float*)d_in[12];
  const float* Who = (const float*)d_in[13];
  const float* bho = (const float*)d_in[14];
  const float* Wxc = (const float*)d_in[15];
  const float* bxc = (const float*)d_in[16];
  const float* Whc = (const float*)d_in[17];
  const float* bhc = (const float*)d_in[18];

  char* ws = (char*)d_ws;
  const size_t MB = 1024 * 1024;
  unsigned short* xb   = (unsigned short*)(ws + 0 * MB);   // [B, IN] bf16   (8 MB)
  unsigned short* hb   = (unsigned short*)(ws + 8 * MB);   // [B, H]  bf16   (8 MB)
  unsigned short* Wxc_ = (unsigned short*)(ws + 16 * MB);  // [4H, IN] bf16  (8 MB)
  unsigned short* Whc_ = (unsigned short*)(ws + 24 * MB);  // [4H, H]  bf16  (8 MB)
  float* bias = (float*)(ws + 32 * MB);                    // [4H] f32

  const int nXH = B_DIM * IN_DIM;  // 4,194,304
  const int nW  = H_DIM * IN_DIM;  // 1,048,576
  dim3 blk(256);

  cvt_f32_to_bf16<<<nXH / 4 / 256, blk, 0, stream>>>(x, xb, nXH / 4);
  cvt_f32_to_bf16<<<nXH / 4 / 256, blk, 0, stream>>>(h, hb, nXH / 4);
  cvt_f32_to_bf16<<<nW / 4 / 256, blk, 0, stream>>>(Wxf, Wxc_ + (size_t)0 * nW, nW / 4);
  cvt_f32_to_bf16<<<nW / 4 / 256, blk, 0, stream>>>(Wxi, Wxc_ + (size_t)1 * nW, nW / 4);
  cvt_f32_to_bf16<<<nW / 4 / 256, blk, 0, stream>>>(Wxo, Wxc_ + (size_t)2 * nW, nW / 4);
  cvt_f32_to_bf16<<<nW / 4 / 256, blk, 0, stream>>>(Wxc, Wxc_ + (size_t)3 * nW, nW / 4);
  cvt_f32_to_bf16<<<nW / 4 / 256, blk, 0, stream>>>(Whf, Whc_ + (size_t)0 * nW, nW / 4);
  cvt_f32_to_bf16<<<nW / 4 / 256, blk, 0, stream>>>(Whi, Whc_ + (size_t)1 * nW, nW / 4);
  cvt_f32_to_bf16<<<nW / 4 / 256, blk, 0, stream>>>(Who, Whc_ + (size_t)2 * nW, nW / 4);
  cvt_f32_to_bf16<<<nW / 4 / 256, blk, 0, stream>>>(Whc, Whc_ + (size_t)3 * nW, nW / 4);
  bias_sum_kernel<<<4, blk, 0, stream>>>(bxf, bhf, bxi, bhi, bxo, bho, bxc, bhc, bias);

  dim3 grid(H_DIM / 16, B_DIM / 256);  // 64 x 16 blocks, 8 waves each
  lstm_wmma_kernel<<<grid, blk, 0, stream>>>(xb, hb, Wxc_, Whc_, bias, c, (float*)d_out);
}
